// AtlasMultiDiffAttn_3066606649538
// MI455X (gfx1250) — matrix operand order repaired and checked by reading, not compiled, not run on hardware
//
#include <hip/hip_runtime.h>

typedef __attribute__((ext_vector_type(16))) _Float16 v16h;
typedef __attribute__((ext_vector_type(8)))  float    v8f;

#define A_LEN 56
#define E_DIM 128
#define LAMBDA_INIT 0.7f
#define LN_EPS 1e-5f

// ---------------- workspace (d_ws) layout, in halves ----------------
// A-fragments for WMMA f32_16x16x32_f16: 32 lanes x 16 halves = 512 halves.
#define WS_EMB   0                          // [t:7][kc:2][mt:4][512]
#define WS_WK    (WS_EMB + 7*2*4*512)       // [kc:4][mt:8][512]
#define WS_ATLAS (WS_WK + 4*8*512)          // [t:7][kc:4][mt:8][512]
#define WS_TOTAL (WS_ATLAS + 7*4*8*512)     // 159,744 halves = 312 KB

// ---------------- LDS layout (bytes) ----------------
// region1 @0     : xT  [134 rows(e+pad)][64 (a)] halves = 17152 B
//                  (reused later as kbuf [56 (a)][136 (o)] halves)
// region2 @17152 : xh  [64 (a)][144 (e)] halves = 18432 B   (alloc 72 rows
//                  = 20736 B; reused later as h1p [72 (a+pad)][144 (e)])
// floats  @37888 : biases / LN params / lambda / qsum / qn / smat / pd
#define SM_R2   17152
#define SM_F    37888
#define SMEM_BYTES (37888 + 1312*4)   // 43136 B

// float-area indices
#define F_BEMB   0     // 64
#define F_BATL   64    // 128
#define F_QNW    192   // 16
#define F_QNB    208
#define F_KNW    224
#define F_KNB    240
#define F_LAM    256
#define F_QSUM   288   // 128
#define F_QN     416   // 128
#define F_SMAT   544   // 8*64
#define F_PD     1056  // 4*64

// =====================================================================
// Weight pre-pack: f32 -> f16 in WMMA A-fragment order.
// A-layout (16-bit A 16x32, ISA 7.12.2): lane L holds row M=L&15;
// half j holds K = j + 16*(L>>4) here (contiguous, not ISA order; B matches).
// =====================================================================
__global__ __launch_bounds__(256) void pack_weights(
    const float* __restrict__ w_emb, const float* __restrict__ w_atlas,
    const float* __restrict__ w_k, _Float16* __restrict__ ws) {
  int idx = blockIdx.x * blockDim.x + threadIdx.x;
  if (idx >= WS_TOTAL) return;
  int within = idx & 511;
  int lane = within >> 4;
  int j = within & 15;
  int mrow = lane & 15;
  int kk = j + 16 * (lane >> 4);
  float v = 0.0f;
  if (idx < WS_WK) {                       // conv1 weights, K = a_in (pad 64)
    int f = idx >> 9;
    int mt = f & 3, kc = (f >> 2) & 1, t = f >> 3;
    int m = mt * 16 + mrow, k = kc * 32 + kk;
    if (m < 56 && k < 56) v = w_emb[(m * 56 + k) * 7 + t];
  } else if (idx < WS_ATLAS) {             // w_k: M = o, K = e
    int f = (idx - WS_WK) >> 9;
    int mt = f & 7, kc = f >> 3;
    int m = mt * 16 + mrow, k = kc * 32 + kk;
    v = w_k[m * 128 + k];
  } else {                                 // conv2 weights: M = e_out, K = e_in
    int f = (idx - WS_ATLAS) >> 9;
    int mt = f & 7, kc = (f >> 3) & 3, t = f >> 5;
    int m = mt * 16 + mrow, k = kc * 32 + kk;
    v = w_atlas[(m * 128 + k) * 7 + t];
  }
  ws[idx] = (_Float16)v;
}

__device__ __forceinline__ float silu_f(float v) {
  return v / (1.0f + __expf(-v));
}

// =====================================================================
// Main kernel: one workgroup (8 wave32) per batch element.
// =====================================================================
__global__ __launch_bounds__(256) void atlas_main(
    const float* __restrict__ x,
    const float* __restrict__ b_emb, const float* __restrict__ b_atlas,
    const float* __restrict__ qn_w, const float* __restrict__ qn_b,
    const float* __restrict__ kn_w, const float* __restrict__ kn_b,
    const float* __restrict__ lq1, const float* __restrict__ lk1,
    const float* __restrict__ lq2, const float* __restrict__ lk2,
    const _Float16* __restrict__ ws, float* __restrict__ out) {
  __shared__ __align__(32) unsigned char smem[SMEM_BYTES];
  _Float16* xT   = (_Float16*)smem;                 // [134][64], rows = e+3
  _Float16* kbuf = (_Float16*)smem;                 // alias: [a][136] (o)
  _Float16* xh   = (_Float16*)(smem + SM_R2);       // [64][144], rows = a
  _Float16* h1p  = xh;                              // alias: [72][144], rows=a+3
  float* fb      = (float*)(smem + SM_F);

  const int b = blockIdx.x;
  const int tid = threadIdx.x;
  const int lane = tid & 31;
  const int wv = tid >> 5;
  const int lhi = lane >> 4;      // 0/1: lane half
  const int llo = lane & 15;

  // ---- Stage A: zero activation regions, stage x (f32->f16) + params ----
  for (int i = tid; i < SM_F / 4; i += 256) ((unsigned int*)smem)[i] = 0u;
  __syncthreads();
  const float* xb = x + (size_t)b * (A_LEN * E_DIM);
  for (int i = tid; i < A_LEN * E_DIM; i += 256) {
    int a = i >> 7, e = i & 127;
    _Float16 h = (_Float16)xb[i];
    xh[a * 144 + e] = h;            // row-major  (k-proj B operand)
    xT[(e + 3) * 64 + a] = h;       // transposed, e-padded (conv1 B operand)
  }
  if (tid < 64)                fb[F_BEMB + tid] = (tid < 56) ? b_emb[tid] : 0.0f;
  else if (tid < 192)          fb[tid] = b_atlas[tid - 64];
  else if (tid < 208)          fb[tid] = qn_w[tid - 192];
  else if (tid < 224)          fb[tid] = qn_b[tid - 208];
  else if (tid < 240)          fb[tid] = kn_w[tid - 224];
  else if (tid < 256)          fb[tid] = kn_b[tid - 240];
  if (tid == 0) {
    float s1 = 0.0f, s2 = 0.0f;
    for (int d = 0; d < 16; ++d) { s1 += lq1[d] * lk1[d]; s2 += lq2[d] * lk2[d]; }
    fb[F_LAM] = __expf(s1) - __expf(s2) + LAMBDA_INIT;
  }
  __syncthreads();

  // ---- Stage B: conv1 WMMAs (per-tap GEMM), results held in VGPRs ----
  // Out tile: M = a_out (4 tiles), N = e (tile = wave id). K = a_in (2 chunks).
  v8f cacc[4];
  {
    const int nt = wv;
#pragma unroll
    for (int mt = 0; mt < 4; ++mt) {
      v8f c = {0.f, 0.f, 0.f, 0.f, 0.f, 0.f, 0.f, 0.f};
#pragma unroll
      for (int t = 0; t < 7; ++t) {
        const _Float16* bp = xT + (nt * 16 + llo + t) * 64 + 16 * lhi;
#pragma unroll
        for (int kc = 0; kc < 2; ++kc) {
          v16h af = *(const v16h*)(ws + WS_EMB + ((t * 2 + kc) * 4 + mt) * 512 + lane * 16);
          v16h bf = *(const v16h*)(bp + kc * 32);
          c = __builtin_amdgcn_wmma_f32_16x16x32_f16(false, af, false, bf,
                                                     (short)0, c, false, false);
        }
      }
      cacc[mt] = c;
    }
  }
  __syncthreads();   // all xT reads done -> region1 reusable as kbuf

  // ---- Stage C: k-projection  kT[o,a] = w_k @ x^T, stored f16 [a][o] ----
  {
    const int mt = wv;   // o tile
#pragma unroll
    for (int nt = 0; nt < 4; ++nt) {
      v8f c = {0.f, 0.f, 0.f, 0.f, 0.f, 0.f, 0.f, 0.f};
      const _Float16* bp = xh + (nt * 16 + llo) * 144 + 16 * lhi;
#pragma unroll
      for (int kc = 0; kc < 4; ++kc) {
        v16h af = *(const v16h*)(ws + WS_WK + (kc * 8 + mt) * 512 + lane * 16);
        v16h bf = *(const v16h*)(bp + kc * 32);
        c = __builtin_amdgcn_wmma_f32_16x16x32_f16(false, af, false, bf,
                                                   (short)0, c, false, false);
      }
      int n_g = nt * 16 + llo;           // a
      if (n_g < 56) {
#pragma unroll
        for (int r = 0; r < 8; ++r) {
          int m_g = mt * 16 + r + 8 * lhi;   // o
          kbuf[n_g * 136 + m_g] = (_Float16)c[r];
        }
      }
    }
  }
  __syncthreads();   // all xh reads done -> region2 reusable as h1p

  // ---- Stage B2: SiLU(conv1 + bias) -> h1p (a-padded, f16) ----
  for (int i = tid; i < 15 * 144; i += 256) {   // zero pad rows 0..2, 59..70
    int rr = i / 144, cc = i - rr * 144;
    int row = (rr < 3) ? rr : (rr + 56);
    h1p[row * 144 + cc] = (_Float16)0.0f;
  }
  {
    const int nt = wv;
#pragma unroll
    for (int mt = 0; mt < 4; ++mt)
#pragma unroll
      for (int r = 0; r < 8; ++r) {
        int m_g = mt * 16 + r + 8 * lhi;       // a_out
        if (m_g < 56) {
          float v = silu_f(cacc[mt][r] + fb[F_BEMB + m_g]);
          h1p[(m_g + 3) * 144 + nt * 16 + llo] = (_Float16)v;
        }
      }
  }
  __syncthreads();

  // ---- Stage D: conv2 WMMAs + SiLU + mean-pool over a -> qsum[e] ----
  // Out tile: M = e_out (tile = wave id), N = a (4 tiles). K = e_in (4 chunks).
  {
    const int mt = wv;
    float qacc[8];
#pragma unroll
    for (int r = 0; r < 8; ++r) qacc[r] = 0.0f;
#pragma unroll
    for (int nt = 0; nt < 4; ++nt) {
      v8f c = {0.f, 0.f, 0.f, 0.f, 0.f, 0.f, 0.f, 0.f};
      int n_g = nt * 16 + llo;                 // a
#pragma unroll
      for (int t = 0; t < 7; ++t) {
        const _Float16* bp = h1p + (n_g + t) * 144 + 16 * lhi;
#pragma unroll
        for (int kc = 0; kc < 4; ++kc) {
          v16h af = *(const v16h*)(ws + WS_ATLAS + ((t * 4 + kc) * 8 + mt) * 512 + lane * 16);
          v16h bf = *(const v16h*)(bp + kc * 32);
          c = __builtin_amdgcn_wmma_f32_16x16x32_f16(false, af, false, bf,
                                                     (short)0, c, false, false);
        }
      }
      if (n_g < 56) {
#pragma unroll
        for (int r = 0; r < 8; ++r) {
          int m_g = mt * 16 + r + 8 * lhi;     // e_out
          qacc[r] += silu_f(c[r] + fb[F_BATL + m_g]);
        }
      }
    }
#pragma unroll
    for (int r = 0; r < 8; ++r) {              // reduce over the 16 a-lanes
      float s = qacc[r];
      s += __shfl_xor(s, 1, 16);
      s += __shfl_xor(s, 2, 16);
      s += __shfl_xor(s, 4, 16);
      s += __shfl_xor(s, 8, 16);
      if (llo == 0) fb[F_QSUM + mt * 16 + r + 8 * lhi] = s * (1.0f / 56.0f);
    }
  }
  __syncthreads();

  // ---- Stage E: q LayerNorm per head group (scale folded in) ----
  if (tid < 8) {
    float m = 0.0f;
    for (int d = 0; d < 16; ++d) m += fb[F_QSUM + tid * 16 + d];
    m *= (1.0f / 16.0f);
    float var = 0.0f;
    for (int d = 0; d < 16; ++d) {
      float t = fb[F_QSUM + tid * 16 + d] - m;
      var += t * t;
    }
    float rs = rsqrtf(var * (1.0f / 16.0f) + LN_EPS);
    for (int d = 0; d < 16; ++d)
      fb[F_QN + tid * 16 + d] =
          ((fb[F_QSUM + tid * 16 + d] - m) * rs * fb[F_QNW + d] + fb[F_QNB + d]) * 0.25f;
  }
  __syncthreads();

  // ---- Stage F: k LayerNorm + attention logits s[h][a] ----
  for (int idx = tid; idx < 56 * 8; idx += 256) {
    int a = idx >> 3, h = idx & 7;
    const _Float16* kp = kbuf + a * 136 + h * 16;
    float v[16], m = 0.0f;
#pragma unroll
    for (int d = 0; d < 16; ++d) { v[d] = (float)kp[d]; m += v[d]; }
    m *= (1.0f / 16.0f);
    float var = 0.0f;
#pragma unroll
    for (int d = 0; d < 16; ++d) { float t = v[d] - m; var += t * t; }
    float rs = rsqrtf(var * (1.0f / 16.0f) + LN_EPS);
    float s = 0.0f;
#pragma unroll
    for (int d = 0; d < 16; ++d)
      s += fb[F_QN + h * 16 + d] * ((v[d] - m) * rs * fb[F_KNW + d] + fb[F_KNB + d]);
    fb[F_SMAT + h * 64 + a] = s;
  }
  __syncthreads();

  // ---- Stage G: softmax over a, per head ----
  if (tid < 8) {
    float* row = fb + F_SMAT + tid * 64;
    float mx = -1e30f;
    for (int a = 0; a < 56; ++a) mx = fmaxf(mx, row[a]);
    float sum = 0.0f;
    for (int a = 0; a < 56; ++a) { row[a] = __expf(row[a] - mx); sum += row[a]; }
    float inv = 1.0f / sum;
    for (int a = 0; a < 56; ++a) row[a] *= inv;
  }
  __syncthreads();

  // ---- Stage H: differential softmax per head-pair ----
  if (tid < 4) {
    float lam = fb[F_LAM];
    float* p0 = fb + F_SMAT + (2 * tid) * 64;
    float* p1 = fb + F_SMAT + (2 * tid + 1) * 64;
    float* pd = fb + F_PD + tid * 64;
    float mx = -1e30f;
    for (int a = 0; a < 56; ++a) {
      float d = p0[a] - lam * p1[a];
      pd[a] = d;
      mx = fmaxf(mx, d);
    }
    float sum = 0.0f;
    for (int a = 0; a < 56; ++a) { pd[a] = __expf(pd[a] - mx); sum += pd[a]; }
    float inv = 1.0f / sum;
    for (int a = 0; a < 56; ++a) pd[a] *= inv;
  }
  __syncthreads();

  // ---- mean over 4 head pairs -> out[b][a] ----
  if (tid < 56)
    out[(size_t)b * 56 + tid] =
        0.25f * (fb[F_PD + tid] + fb[F_PD + 64 + tid] +
                 fb[F_PD + 128 + tid] + fb[F_PD + 192 + tid]);
}

extern "C" void kernel_launch(void* const* d_in, const int* in_sizes, int n_in,
                              void* d_out, int out_size, void* d_ws, size_t ws_size,
                              hipStream_t stream) {
  const float* x       = (const float*)d_in[0];
  const float* w_emb   = (const float*)d_in[1];
  const float* b_emb   = (const float*)d_in[2];
  const float* w_atlas = (const float*)d_in[3];
  const float* b_atlas = (const float*)d_in[4];
  const float* w_k     = (const float*)d_in[5];
  const float* qn_w    = (const float*)d_in[6];
  const float* qn_b    = (const float*)d_in[7];
  const float* kn_w    = (const float*)d_in[8];
  const float* kn_b    = (const float*)d_in[9];
  const float* lq1     = (const float*)d_in[10];
  const float* lk1     = (const float*)d_in[11];
  const float* lq2     = (const float*)d_in[12];
  const float* lk2     = (const float*)d_in[13];
  _Float16* ws = (_Float16*)d_ws;
  int B = in_sizes[0] / (A_LEN * E_DIM);

  pack_weights<<<(WS_TOTAL + 255) / 256, 256, 0, stream>>>(w_emb, w_atlas, w_k, ws);
  atlas_main<<<B, 256, 0, stream>>>(x, b_emb, b_atlas, qn_w, qn_b, kn_w, kn_b,
                                    lq1, lk1, lq2, lk2, ws, (float*)d_out);
}
